// GraphConnectivityAnalyzer_73366631350910
// MI455X (gfx1250) — matrix operand order repaired and checked by reading, not compiled, not run on hardware
//
#include <hip/hip_runtime.h>
#include <hip/hip_bf16.h>

typedef __attribute__((ext_vector_type(16))) _Float16 v16h;
typedef __attribute__((ext_vector_type(8)))  float    v8f;
typedef __attribute__((ext_vector_type(2)))  float    v2f;

#define BATCH 32
#define CHAN  64
#define TLEN  8192
#define EMB   64

union F16Frag { uint4 q[2]; v16h v; };

// ---------------------------------------------------------------------------
// Kernel 1: Hilbert analytic phasor per (b,c) row.
// 8192-pt radix-2 DIT FFT in LDS (64KB), hilbert multiplier, inverse FFT,
// normalize to unit phasors, store fp16 Re/Im planes (1/T scale cancels).
// ---------------------------------------------------------------------------
__global__ __launch_bounds__(256) void hilbert_phasor_kernel(const float* __restrict__ x,
                                                             _Float16* __restrict__ zr,
                                                             _Float16* __restrict__ zi)
{
    extern __shared__ float smem1[];
    float* re = smem1;          // 8192 f32
    float* im = smem1 + TLEN;   // 8192 f32
    const int tid = threadIdx.x;
    const size_t row = blockIdx.x;  // b*C + c
    const float* xrow = x + row * TLEN;

    for (int i = tid; i < TLEN; i += 256) { re[i] = xrow[i]; im[i] = 0.f; }
    __syncthreads();

    for (int pass = 0; pass < 2; ++pass) {
        const float sgn = (pass == 0) ? 1.f : -1.f;  // pass0 = forward (e^{-i..}), pass1 = inverse
        // bit-reversal permutation (13 bits)
        for (int i = tid; i < TLEN; i += 256) {
            const int j = (int)(__builtin_bitreverse32((unsigned)i) >> 19);
            if (j > i) {
                float t;
                t = re[i]; re[i] = re[j]; re[j] = t;
                t = im[i]; im[i] = im[j]; im[j] = t;
            }
        }
        __syncthreads();
        for (int s = 1; s <= 13; ++s) {
            const int half = 1 << (s - 1);
            for (int u = tid; u < (TLEN >> 1); u += 256) {
                const int pos = u & (half - 1);
                const int i0  = ((u >> (s - 1)) << s) + pos;
                const int i1  = i0 + half;
                const float ang = sgn * -6.28318530717958647692f * (float)pos / (float)(2 * half);
                float wi, wr;
                __sincosf(ang, &wi, &wr);
                const float br = re[i1], bi = im[i1];
                const float tr = wr * br - wi * bi;
                const float ti = wr * bi + wi * br;
                const float ar = re[i0], ai = im[i0];
                re[i1] = ar - tr; im[i1] = ai - ti;
                re[i0] = ar + tr; im[i0] = ai + ti;
            }
            __syncthreads();
        }
        if (pass == 0) {
            for (int i = tid; i < TLEN; i += 256) {
                const float m = (i == 0 || i == TLEN / 2) ? 1.f : (i < TLEN / 2 ? 2.f : 0.f);
                re[i] *= m; im[i] *= m;
            }
            __syncthreads();
        }
    }

    _Float16* zrow_r = zr + row * TLEN;
    _Float16* zrow_i = zi + row * TLEN;
    for (int i = tid; i < TLEN; i += 256) {
        const float a = re[i], b = im[i];
        const float inv = rsqrtf(fmaxf(a * a + b * b, 1e-30f));
        zrow_r[i] = (_Float16)(a * inv);
        zrow_i[i] = (_Float16)(b * inv);
    }
}

// ---------------------------------------------------------------------------
// Kernel 2: PLV connectivity = |Z Z^H| / T via f16 WMMA, f32 accumulation.
// grid = B*16 (one block per 16x16 output tile); 8 waves split K=8192.
// A frag (16x32 f16): lane M=lane&15, half=lane>>4; elems 0-7 at K=k0+half*8,
// elems 8-15 at K=k0+16+half*8 (two contiguous 16B runs).
// B frag (32x16 f16): lane N=lane&15, same K order as A (k0+half*8, k0+16+half*8).
// ---------------------------------------------------------------------------
__device__ inline v8f wmma_f16(v16h a, v16h b, v8f c) {
    return __builtin_amdgcn_wmma_f32_16x16x32_f16(false, a, false, b, (short)0, c, false, false);
}

__global__ __launch_bounds__(256) void plv_gram_kernel(const _Float16* __restrict__ zr,
                                                       const _Float16* __restrict__ zi,
                                                       float* __restrict__ conn)
{
    const int blk  = blockIdx.x;
    const int b    = blk >> 4;
    const int tile = blk & 15;
    const int tm = tile >> 2, tn = tile & 3;
    const int tid  = threadIdx.x;
    const int wave = tid >> 5, lane = tid & 31;
    const int l15 = lane & 15, half = lane >> 4;

    const _Float16* zr_b = zr + (size_t)b * CHAN * TLEN;
    const _Float16* zi_b = zi + (size_t)b * CHAN * TLEN;
    const _Float16* arow_r = zr_b + (size_t)(tm * 16 + l15) * TLEN;
    const _Float16* arow_i = zi_b + (size_t)(tm * 16 + l15) * TLEN;
    const _Float16* brow_r = zr_b + (size_t)(tn * 16 + l15) * TLEN;
    const _Float16* brow_i = zi_b + (size_t)(tn * 16 + l15) * TLEN;

    v8f acc_re = {};   // sum ArBr + AiBi
    v8f acc_x  = {};   // sum AiBr
    v8f acc_y  = {};   // sum ArBi   (Im = acc_x - acc_y; f16 WMMA has no A/B neg)

    const int k_begin = wave * (TLEN / 8);
    const int k_end   = k_begin + (TLEN / 8);
    for (int k0 = k_begin; k0 < k_end; k0 += 32) {
        __builtin_prefetch((const void*)(arow_r + k0 + 512), 0, 1);
        __builtin_prefetch((const void*)(brow_r + k0 + 512), 0, 1);
        F16Frag ar, ai, br, bi;
        ar.q[0] = *(const uint4*)(arow_r + k0 + half * 8);
        ar.q[1] = *(const uint4*)(arow_r + k0 + 16 + half * 8);
        ai.q[0] = *(const uint4*)(arow_i + k0 + half * 8);
        ai.q[1] = *(const uint4*)(arow_i + k0 + 16 + half * 8);
        br.q[0] = *(const uint4*)(brow_r + k0 + half * 8);
        br.q[1] = *(const uint4*)(brow_r + k0 + 16 + half * 8);
        bi.q[0] = *(const uint4*)(brow_i + k0 + half * 8);
        bi.q[1] = *(const uint4*)(brow_i + k0 + 16 + half * 8);
        acc_re = wmma_f16(ar.v, br.v, acc_re);
        acc_re = wmma_f16(ai.v, bi.v, acc_re);
        acc_x  = wmma_f16(ai.v, br.v, acc_x);
        acc_y  = wmma_f16(ar.v, bi.v, acc_y);
    }

    __shared__ float red[3][8][256];
    for (int r = 0; r < 8; ++r) {
        red[0][wave][r * 32 + lane] = acc_re[r];
        red[1][wave][r * 32 + lane] = acc_x[r];
        red[2][wave][r * 32 + lane] = acc_y[r];
    }
    __syncthreads();

    float sre = 0.f, sx = 0.f, sy = 0.f;
    for (int w = 0; w < 8; ++w) {
        sre += red[0][w][tid];
        sx  += red[1][w][tid];
        sy  += red[2][w][tid];
    }
    const float sim = sx - sy;
    float plv = sqrtf(sre * sre + sim * sim) * (1.0f / (float)TLEN);
    const int r = tid >> 5, l = tid & 31;
    const int row = tm * 16 + ((l >> 4) * 8 + r);
    const int col = tn * 16 + (l & 15);
    if (row == col) plv = 0.f;
    conn[((size_t)b * CHAN + row) * CHAN + col] = plv;
}

// ---------------------------------------------------------------------------
// Kernel 3: GCN (2 layers) + 8-head MHA per batch, matmuls on f32 WMMA 16x16x4.
// A frag (16x4 f32): M=lane&15, K = (lane>>4)*2 + r.
// B frag (4x16 f32): N=lane&15, K = (lane>>4)*2 + r.
// C/D frag: row = (lane>>4)*8 + r, col = lane&15.
// ---------------------------------------------------------------------------
__device__ void gemm_wmma_f32(const float* __restrict__ A, int lda,
                              const float* __restrict__ B, int ldb, bool transB,
                              float* __restrict__ C, int ldc,
                              int M, int N, int K,
                              const float* __restrict__ bias, bool relu, int tid)
{
    const int wave = tid >> 5, lane = tid & 31;
    const int l15 = lane & 15, h2v = lane >> 4;
    const int tilesN = N >> 4;
    const int tiles  = (M >> 4) * tilesN;
    for (int t = wave; t < tiles; t += 8) {
        const int tm = t / tilesN, tn = t % tilesN;
        const int n = tn * 16 + l15;
        v8f acc = {};
        if (bias) { const float bv = bias[n]; for (int r = 0; r < 8; ++r) acc[r] = bv; }
        const float* arow = A + (size_t)(tm * 16 + l15) * lda;
        const float* brow = transB ? (B + (size_t)n * ldb) : nullptr;
        for (int k0 = 0; k0 < K; k0 += 4) {
            v2f a, bv2;
            a[0] = arow[k0 + h2v * 2 + 0];
            a[1] = arow[k0 + h2v * 2 + 1];
            if (transB) {
                bv2[0] = brow[k0 + h2v * 2 + 0];
                bv2[1] = brow[k0 + h2v * 2 + 1];
            } else {
                bv2[0] = B[(size_t)(k0 + h2v * 2 + 0) * ldb + n];
                bv2[1] = B[(size_t)(k0 + h2v * 2 + 1) * ldb + n];
            }
            acc = __builtin_amdgcn_wmma_f32_16x16x4_f32(false, a, false, bv2,
                                                        (short)0, acc, false, false);
        }
        for (int r = 0; r < 8; ++r) {
            float v = acc[r];
            if (relu) v = fmaxf(v, 0.f);
            C[(size_t)(tm * 16 + h2v * 8 + r) * ldc + n] = v;
        }
    }
}

__global__ __launch_bounds__(256) void gcn_mha_kernel(const float* __restrict__ conn_g,
                                                      const float* __restrict__ emb,
                                                      const float* __restrict__ w1,
                                                      const float* __restrict__ b1,
                                                      const float* __restrict__ w2,
                                                      const float* __restrict__ b2,
                                                      const float* __restrict__ in_w,
                                                      const float* __restrict__ in_b,
                                                      const float* __restrict__ out_w,
                                                      const float* __restrict__ out_b,
                                                      float* __restrict__ gf_out,
                                                      float* __restrict__ h2_out)
{
    extern __shared__ float smem3[];
    float* s_conn = smem3;            // 64x64
    float* s_h2   = smem3 + 4096;     // 64x64
    float* s_t1   = smem3 + 8192;     // 64x128
    float* s_h1   = smem3 + 16384;    // 64x128
    float* s_t2   = smem3 + 24576;    // 64x64
    float* s_qkv  = smem3 + 28672;    // 64x192
    float* s_att  = smem3 + 40960;    // 64x64 per-head scores
    float* s_ctx  = smem3 + 45056;    // 64x64 concat context

    const int tid = threadIdx.x;
    const int b   = blockIdx.x;

    for (int i = tid; i < 4096; i += 256) s_conn[i] = conn_g[(size_t)b * 4096 + i];
    __syncthreads();

    // t1 = emb @ w1  (64x64 @ 64x128)
    gemm_wmma_f32(emb, EMB, w1, 2 * EMB, false, s_t1, 2 * EMB, 64, 128, 64, nullptr, false, tid);
    __syncthreads();
    // h1 = relu(conn @ t1 + b1)
    gemm_wmma_f32(s_conn, 64, s_t1, 2 * EMB, false, s_h1, 2 * EMB, 64, 128, 64, b1, true, tid);
    __syncthreads();
    // t2 = h1 @ w2  (64x128 @ 128x64)
    gemm_wmma_f32(s_h1, 2 * EMB, w2, EMB, false, s_t2, EMB, 64, 64, 128, nullptr, false, tid);
    __syncthreads();
    // h2 = conn @ t2 + b2
    gemm_wmma_f32(s_conn, 64, s_t2, EMB, false, s_h2, EMB, 64, 64, 64, b2, false, tid);
    __syncthreads();
    for (int i = tid; i < 4096; i += 256) h2_out[(size_t)b * 4096 + i] = s_h2[i];

    // qkv = h2 @ in_proj_w^T + in_proj_b   (64x64 @ 64x192)
    gemm_wmma_f32(s_h2, EMB, in_w, EMB, true, s_qkv, 3 * EMB, 64, 192, 64, in_b, false, tid);
    __syncthreads();

    // 8 heads, D = 8: scores -> softmax -> context (VALU; K=8 too small for WMMA)
    const float scale = 0.35355339059327373f;  // 1/sqrt(8)
    for (int hh = 0; hh < 8; ++hh) {
        for (int idx = tid; idx < 4096; idx += 256) {
            const int qr = idx >> 6, kc = idx & 63;
            float dot = 0.f;
            for (int d = 0; d < 8; ++d)
                dot += s_qkv[qr * 192 + hh * 8 + d] * s_qkv[kc * 192 + 64 + hh * 8 + d];
            s_att[idx] = dot * scale;
        }
        __syncthreads();
        if (tid < 64) {
            float* rowp = s_att + tid * 64;
            float mx = rowp[0];
            for (int j = 1; j < 64; ++j) mx = fmaxf(mx, rowp[j]);
            float sum = 0.f;
            for (int j = 0; j < 64; ++j) { const float e = __expf(rowp[j] - mx); rowp[j] = e; sum += e; }
            const float inv = 1.f / sum;
            for (int j = 0; j < 64; ++j) rowp[j] *= inv;
        }
        __syncthreads();
        for (int idx = tid; idx < 512; idx += 256) {
            const int rr = idx >> 3, d = idx & 7;
            float dot = 0.f;
            for (int k = 0; k < 64; ++k)
                dot += s_att[rr * 64 + k] * s_qkv[k * 192 + 128 + hh * 8 + d];
            s_ctx[rr * 64 + hh * 8 + d] = dot;
        }
        __syncthreads();
    }

    // graph_features = ctx @ out_proj_w^T + out_proj_b  -> global
    gemm_wmma_f32(s_ctx, EMB, out_w, EMB, true, gf_out + (size_t)b * 4096, EMB,
                  64, 64, 64, out_b, false, tid);
}

// ---------------------------------------------------------------------------
extern "C" void kernel_launch(void* const* d_in, const int* in_sizes, int n_in,
                              void* d_out, int out_size, void* d_ws, size_t ws_size,
                              hipStream_t stream) {
    const float* x     = (const float*)d_in[0];
    const float* emb   = (const float*)d_in[1];
    const float* w1    = (const float*)d_in[2];
    const float* b1    = (const float*)d_in[3];
    const float* w2    = (const float*)d_in[4];
    const float* b2    = (const float*)d_in[5];
    const float* in_w  = (const float*)d_in[6];
    const float* in_b  = (const float*)d_in[7];
    const float* out_w = (const float*)d_in[8];
    const float* out_b = (const float*)d_in[9];

    float* out = (float*)d_out;
    float* conn_out = out;                       // B*C*C
    float* gf_out   = out + BATCH * CHAN * CHAN; // B*N*E
    float* h2_out   = out + 2 * BATCH * CHAN * CHAN;

    _Float16* zr = (_Float16*)d_ws;
    _Float16* zi = zr + (size_t)BATCH * CHAN * TLEN;

    hilbert_phasor_kernel<<<BATCH * CHAN, 256, 2 * TLEN * sizeof(float), stream>>>(x, zr, zi);
    plv_gram_kernel<<<BATCH * 16, 256, 0, stream>>>(zr, zi, conn_out);
    gcn_mha_kernel<<<BATCH, 256, 49152 * sizeof(float), stream>>>(
        conn_out, emb, w1, b1, w2, b2, in_w, in_b, out_w, out_b, gf_out, h2_out);
}
